// model_38388417691745
// MI455X (gfx1250) — compile-verified, matrix operand order checked by reading, not run on hardware
//
#include <hip/hip_runtime.h>
#include <hip/hip_bf16.h>

// ---------------------------------------------------------------------------
// GNN forward for MI455X (gfx1250, wave32).
// Dense layers: v_wmma_f32_16x16x32_f16, f16 A/B (converted from f32 in
// flight), f32 accumulate.  All GEMMs have M%16==0, N%32==0, K%32==0: zero
// per-lane guards.  A: 4x global_load_b128 + cvt_pk.  B: one float4 per
// thread staged to LDS in fragment order (fixed per-thread row/col =>
// pointer bump of 32*ldb per K-step, no per-element address math); each lane
// then reads its whole B fragment with 2x ds_load_b128.
// ECC layer-2 GEMM fused with einsum + recv-scatter (saves the E x 384
// intermediate, ~1.2 GB of HBM round-trip).
// Workspace requirement: 3 * N * 512 * 4B + ~5 MB  (~312 MB).
// ---------------------------------------------------------------------------

typedef __attribute__((ext_vector_type(16))) _Float16 v16h;
typedef __attribute__((ext_vector_type(8)))  float    v8f;

#define GW 8            // waves per block for WMMA kernels
#define TB (GW * 32)    // 256 threads
#define TILE_K 32

// Fragment order: lane l holds row/col (l&15), half hi=l>>4; element e maps
// to k = hi*8+e (e<8) or 16+hi*8+(e-8).  Inverse (k -> lane-half, e):
//   hi = (k>>3)&1 ;  e = ((k>>4)<<3) | (k&7)
__device__ __forceinline__ void frag_from_k(int k, int& hi, int& e) {
    hi = (k >> 3) & 1;
    e  = ((k >> 4) << 3) | (k & 7);
}

// Build an A fragment from a row-major f32 row: two contiguous 8-float runs.
__device__ __forceinline__ v16h load_fragA(const float* __restrict__ arow,
                                           int k0, int hi) {
    const float* ap = arow + k0 + hi * 8;
    float t[16];
    *(float4*)(t + 0)  = *(const float4*)(ap + 0);
    *(float4*)(t + 4)  = *(const float4*)(ap + 4);
    *(float4*)(t + 8)  = *(const float4*)(ap + 16);
    *(float4*)(t + 12) = *(const float4*)(ap + 20);
    v16h f;
#pragma unroll
    for (int e = 0; e < 16; ++e) f[e] = (_Float16)t[e];
    return f;
}

// ---------------------------------------------------------------------------
// Generic WMMA GEMM: D(MxN) = epilogue(A(MxK) @ B(KxN) + bias)
// Requires: M%16==0, N%32==0, K%32==0.
// Wave computes a 16x32 tile (one A frag, two B frags, two WMMAs/K-step).
// mode 0: identity (+bias)   mode 1: relu(+bias)
// mode 2: leaky(0.15) then BN affine g*(y-m)*rsqrt(v+1e-3)+b
// ---------------------------------------------------------------------------
__global__ __launch_bounds__(TB)
void wmma_gemm_kernel(const float* __restrict__ A, int lda,
                      const float* __restrict__ B, int ldb,
                      float* __restrict__ D, int ldd,
                      int M, int N, int K,
                      const float* __restrict__ bias, int mode,
                      const float* __restrict__ bng, const float* __restrict__ bnb,
                      const float* __restrict__ bnm, const float* __restrict__ bnv) {
    __shared__ alignas(32) _Float16 Bs[2][2 * 512];   // [buf][sub*512 + lane*16 + e]
    const int tid  = threadIdx.x;
    const int lane = tid & 31;
    const int wave = tid >> 5;
    const int hi   = lane >> 4;
    const int m0   = (blockIdx.x * GW + wave) * 16;
    const int n0   = blockIdx.y * 32;
    const bool active = (m0 < M);
    const int mrow = m0 + (lane & 15);
    const float* arow = A + (size_t)mrow * lda;

    // --- per-thread constant staging geometry: one float4 of B per K-step ---
    const int skq = tid >> 3;                 // tile row k (0..31), fixed
    const int sq  = tid & 7;                  // float4 column group (0..7)
    int bhi, be; frag_from_k(skq, bhi, be);
    const int sbase = 256 * bhi + be;         // + nn*16 + sub*512
    const float* bptr = B + (size_t)skq * ldb + n0 + sq * 4;

    v8f acc0 = {}, acc1 = {};
    const int nk = K / TILE_K;
    for (int kt = 0; kt < nk; ++kt) {
        const int k0 = kt * TILE_K;
        const int p  = kt & 1;
        {   // stage B[k0:k0+32, n0:n0+32] into fragment order
            const float4 bv = *(const float4*)bptr;
            bptr += (size_t)32 * ldb;
#pragma unroll
            for (int j = 0; j < 4; ++j) {
                const int nc = sq * 4 + j;
                const int sub = nc >> 4, nn = nc & 15;
                Bs[p][sub * 512 + nn * 16 + sbase] =
                    (_Float16)((const float*)&bv)[j];
            }
        }
        __syncthreads();
        if (active) {
            const v16h fa  = load_fragA(arow, k0, hi);
            const v16h fb0 = *(const v16h*)(&Bs[p][0 * 512 + lane * 16]);
            const v16h fb1 = *(const v16h*)(&Bs[p][1 * 512 + lane * 16]);
            acc0 = __builtin_amdgcn_wmma_f32_16x16x32_f16(
                false, fa, false, fb0, (short)0, acc0, false, false);
            acc1 = __builtin_amdgcn_wmma_f32_16x16x32_f16(
                false, fa, false, fb1, (short)0, acc1, false, false);
        }
    }

    if (!active) return;
#pragma unroll
    for (int sub = 0; sub < 2; ++sub) {
        const int n = n0 + sub * 16 + (lane & 15);
        const v8f& acc = sub ? acc1 : acc0;
        const float bv = bias ? bias[n] : 0.f;
        float sc = 1.f, sh = 0.f;
        if (mode == 2) {
            sc = bng[n] * rsqrtf(bnv[n] + 1.0e-3f);
            sh = bnb[n] - sc * bnm[n];
        }
#pragma unroll
        for (int v = 0; v < 8; ++v) {
            const int m = m0 + v + 8 * hi;
            float val = acc[v] + bv;
            if (mode == 1)      val = fmaxf(val, 0.f);
            else if (mode == 2) val = ((val >= 0.f) ? val : 0.15f * val) * sc + sh;
            D[(size_t)m * ldd + n] = val;
        }
    }
}

// ---------------------------------------------------------------------------
// Fused ECC layer-2 + einsum + recv-scatter:
//   k3[e, f*64+c] = k2[e,:] @ W2[:, f*64+c] + b2
//   msg[e,c]      = sum_f x[send[e], f] * k3[e, f*64+c]
//   h_sum[recv[e], c] += msg[e, c]
// Wave owns 16 edges x 16 c-cols; loops 6 f-groups x 2 K-steps (K=64).
// All W2 offsets fold into immediates off one base pointer.
// ---------------------------------------------------------------------------
__global__ __launch_bounds__(TB)
void ecc_msg_kernel(const float* __restrict__ K2,   // E x 64
                    const float* __restrict__ W2,   // 64 x 384
                    const float* __restrict__ B2,   // 384
                    const float* __restrict__ X,    // N x 6
                    const int*   __restrict__ EI,   // E x 2
                    float* __restrict__ Hsum,       // N x 64
                    int E) {
    __shared__ alignas(32) _Float16 Bs[2][512];
    const int tid  = threadIdx.x;
    const int lane = tid & 31;
    const int wave = tid >> 5;
    const int hi   = lane >> 4;
    const int m0   = (blockIdx.x * GW + wave) * 16;  // edge tile
    const int c0   = blockIdx.y * 16;                // c tile (0..63)
    const bool active = (m0 < E);
    const int mrow = m0 + (lane & 15);
    const float* arow = K2 + (size_t)mrow * 64;

    // per-thread constant staging geometry: one float2 of W2 per stage
    const int skq = tid >> 3;                 // tile row k (0..31), fixed
    const int sq  = tid & 7;                  // float2 col group (0..7)
    int bhi, be; frag_from_k(skq, bhi, be);
    const int sbase = 256 * bhi + be;
    const float* wbase = W2 + (size_t)skq * 384 + c0 + sq * 2;

    // hoist edge indices for this wave's 8 rows
    int sidx[8], ridx[8];
    if (active) {
#pragma unroll
        for (int v = 0; v < 8; ++v) {
            const int m = m0 + v + 8 * hi;
            sidx[v] = EI[2 * m];
            ridx[v] = EI[2 * m + 1];
        }
    }

    float macc[8] = {0.f, 0.f, 0.f, 0.f, 0.f, 0.f, 0.f, 0.f};
    int stage = 0;
#pragma unroll
    for (int f = 0; f < 6; ++f) {
        v8f acc = {};
        const int nbase = f * 64;
#pragma unroll
        for (int kt = 0; kt < 2; ++kt) {            // K = 64
            const int k0 = kt * 32;
            const int p  = stage & 1;
            ++stage;
            {
                const float2 bv = *(const float2*)(wbase + nbase + (size_t)k0 * 384);
#pragma unroll
                for (int j = 0; j < 2; ++j) {
                    const int nn = sq * 2 + j;      // 0..15
                    Bs[p][nn * 16 + sbase] = (_Float16)((const float*)&bv)[j];
                }
            }
            __syncthreads();
            if (active) {
                const v16h fa = load_fragA(arow, k0, hi);
                const v16h fb = *(const v16h*)(&Bs[p][lane * 16]);
                acc = __builtin_amdgcn_wmma_f32_16x16x32_f16(
                    false, fa, false, fb, (short)0, acc, false, false);
            }
        }
        if (active) {
            const int n  = nbase + c0 + (lane & 15);
            const float bv = B2[n];
#pragma unroll
            for (int v = 0; v < 8; ++v) {
                macc[v] += (acc[v] + bv) * X[(size_t)sidx[v] * 6 + f];
            }
        }
    }
    if (!active) return;
    const int c = c0 + (lane & 15);
#pragma unroll
    for (int v = 0; v < 8; ++v) {
        atomicAdd(&Hsum[(size_t)ridx[v] * 64 + c], macc[v]);
    }
}

// ---------------------------------------------------------------------------
// Elementwise / scatter helpers
// ---------------------------------------------------------------------------
__global__ void zero_kernel(float* __restrict__ p, long long n) {
    long long t = (long long)blockIdx.x * blockDim.x + threadIdx.x;
    if (t < n) p[t] = 0.f;
}

// Pad ecc_w0 (7x64) to 32x64 with zero rows so GEMM0 needs no K guard.
__global__ void pad_w0_kernel(const float* __restrict__ w0,
                              float* __restrict__ wp) {
    int t = blockIdx.x * blockDim.x + threadIdx.x;
    if (t >= 32 * 64) return;
    const int k = t >> 6, n = t & 63;
    wp[t] = (k < 7) ? w0[k * 64 + n] : 0.f;
}

__global__ void edge_feat_kernel(const float* __restrict__ X,
                                 const int* __restrict__ EI,
                                 float* __restrict__ Ef,   // E x 32 (padded)
                                 float* __restrict__ cnt,  // N
                                 int E) {
    int e = blockIdx.x * blockDim.x + threadIdx.x;
    if (e >= E) return;
    const int s = EI[2 * e], r = EI[2 * e + 1];
    float d[6];
#pragma unroll
    for (int f = 0; f < 6; ++f) d[f] = X[(size_t)r * 6 + f] - X[(size_t)s * 6 + f];
    const float dist = sqrtf(d[0]*d[0] + d[1]*d[1] + d[2]*d[2]);
    const float inv  = (dist == 0.f) ? 0.f : 1.f / dist;   // divide_no_nan
    float* o = Ef + (size_t)e * 32;
    o[0] = d[3]; o[1] = d[4]; o[2] = d[5]; o[3] = dist;
    o[4] = d[0] * inv; o[5] = d[1] * inv; o[6] = d[2] * inv;
#pragma unroll
    for (int j = 7; j < 32; ++j) o[j] = 0.f;
    atomicAdd(&cnt[r], 1.0f);
}

__global__ void ecc_combine_kernel(const float* __restrict__ Hsum,
                                   const float* __restrict__ cnt,
                                   const float* __restrict__ X,
                                   const float* __restrict__ root,  // 6 x 64
                                   const float* __restrict__ bias,  // 64
                                   float* __restrict__ H, int Nn) {
    int t = blockIdx.x * blockDim.x + threadIdx.x;
    if (t >= Nn * 64) return;
    const int n = t >> 6, c = t & 63;
    float acc = Hsum[t] / fmaxf(cnt[n], 1.f);
    const float* xr = X + (size_t)n * 6;
#pragma unroll
    for (int f = 0; f < 6; ++f) acc += xr[f] * root[f * 64 + c];
    acc += bias[c];
    H[t] = fmaxf(acc, 0.f);
}

// agg[send[e], c] += hw[recv[e], c]
__global__ void gcn_scatter_kernel(const float* __restrict__ HW,
                                   const int* __restrict__ EI,
                                   float* __restrict__ Agg,
                                   int E, int cshift) {
    long long t = (long long)blockIdx.x * blockDim.x + threadIdx.x;
    if (t >= ((long long)E << cshift)) return;
    const int e = (int)(t >> cshift);
    const int c = (int)(t & ((1 << cshift) - 1));
    const int s = EI[2 * e], r = EI[2 * e + 1];
    atomicAdd(&Agg[((size_t)s << cshift) + c], HW[((size_t)r << cshift) + c]);
}

__global__ void bias_relu_kernel(float* __restrict__ Agg,
                                 const float* __restrict__ b,
                                 int Nn, int cshift) {
    long long t = (long long)blockIdx.x * blockDim.x + threadIdx.x;
    if (t >= ((long long)Nn << cshift)) return;
    const int c = (int)(t & ((1 << cshift) - 1));
    Agg[t] = fmaxf(Agg[t] + b[c], 0.f);
}

// segment max / sum / count over graph ids (post-ReLU values >= 0, so uint
// bit-pattern atomicMax on floats is order-preserving; gmax init 0).
__global__ void pool_kernel(const float* __restrict__ H4,   // N x 512
                            const int* __restrict__ gi,     // N
                            float* __restrict__ gsum,       // 256 x 512
                            unsigned* __restrict__ gmaxb,   // 256 x 512
                            float* __restrict__ gcnt,       // 256
                            int Nn) {
    long long t = (long long)blockIdx.x * blockDim.x + threadIdx.x;
    if (t >= ((long long)Nn << 9)) return;
    const int n = (int)(t >> 9), c = (int)(t & 511);
    const int g = gi[n];
    const float v = H4[t];
    atomicAdd(&gsum[(size_t)g * 512 + c], v);
    atomicMax(&gmaxb[(size_t)g * 512 + c], __float_as_uint(v));
    if (c == 0) atomicAdd(&gcnt[g], 1.f);
}

__global__ void zbuild_kernel(const float* __restrict__ gsum,
                              const unsigned* __restrict__ gmaxb,
                              const float* __restrict__ gcnt,
                              float* __restrict__ Z) {  // 256 x 1024
    int t = blockIdx.x * blockDim.x + threadIdx.x;
    if (t >= 256 * 1024) return;
    const int g = t >> 10, c = t & 1023;
    float v;
    if (c < 512) v = __uint_as_float(gmaxb[(size_t)g * 512 + c]);
    else         v = gsum[(size_t)g * 512 + (c - 512)] / fmaxf(gcnt[g], 1.f);
    Z[t] = v;
}

// u = normalize((A1 @ Wout + Bout)); out = [u(2) | abs(s)+eps (64)]
__global__ void final_kernel(const float* __restrict__ A1,   // 256 x 64
                             const float* __restrict__ Wout, // 64 x 2
                             const float* __restrict__ Bout, // 2
                             const float* __restrict__ S1,   // 256 x 64
                             float* __restrict__ Out) {      // 256 x 66
    const int row = blockIdx.x;
    const int c   = threadIdx.x;   // 64 threads
    Out[(size_t)row * 66 + 2 + c] = fabsf(S1[(size_t)row * 64 + c]) + 1.0e-5f;
    if (c == 0) {
        float u0 = Bout[0], u1 = Bout[1];
        const float* ar = A1 + (size_t)row * 64;
        for (int k = 0; k < 64; ++k) {
            const float a = ar[k];
            u0 += a * Wout[k * 2 + 0];
            u1 += a * Wout[k * 2 + 1];
        }
        const float nrm = sqrtf(u0 * u0 + u1 * u1);
        const float inv = (nrm == 0.f) ? 0.f : 1.f / nrm;  // divide_no_nan
        Out[(size_t)row * 66 + 0] = u0 * inv;
        Out[(size_t)row * 66 + 1] = u1 * inv;
    }
}

// ---------------------------------------------------------------------------
extern "C" void kernel_launch(void* const* d_in, const int* in_sizes, int n_in,
                              void* d_out, int out_size, void* d_ws, size_t ws_size,
                              hipStream_t stream) {
    const float* x    = (const float*)d_in[0];
    const int*   ei   = (const int*)d_in[1];
    const int*   gi   = (const int*)d_in[2];
    const float* w0   = (const float*)d_in[3];
    const float* b0   = (const float*)d_in[4];
    const float* w1   = (const float*)d_in[5];
    const float* b1   = (const float*)d_in[6];
    const float* w2   = (const float*)d_in[7];
    const float* b2   = (const float*)d_in[8];
    const float* root = (const float*)d_in[9];
    const float* rb   = (const float*)d_in[10];
    const float* g1w  = (const float*)d_in[11]; const float* g1b = (const float*)d_in[12];
    const float* g2w  = (const float*)d_in[13]; const float* g2b = (const float*)d_in[14];
    const float* g3w  = (const float*)d_in[15]; const float* g3b = (const float*)d_in[16];
    const float* g4w  = (const float*)d_in[17]; const float* g4b = (const float*)d_in[18];
    const float* d0w = (const float*)d_in[19]; const float* d0b = (const float*)d_in[20];
    const float* bn0g = (const float*)d_in[21]; const float* bn0b = (const float*)d_in[22];
    const float* bn0m = (const float*)d_in[23]; const float* bn0v = (const float*)d_in[24];
    const float* d1w = (const float*)d_in[25]; const float* d1b = (const float*)d_in[26];
    const float* bn1g = (const float*)d_in[27]; const float* bn1b = (const float*)d_in[28];
    const float* bn1m = (const float*)d_in[29]; const float* bn1v = (const float*)d_in[30];
    const float* d2w = (const float*)d_in[31]; const float* d2b = (const float*)d_in[32];
    const float* bn2g = (const float*)d_in[33]; const float* bn2b = (const float*)d_in[34];
    const float* bn2m = (const float*)d_in[35]; const float* bn2v = (const float*)d_in[36];
    const float* a0w = (const float*)d_in[37]; const float* a0b = (const float*)d_in[38];
    const float* a1w = (const float*)d_in[39]; const float* a1b = (const float*)d_in[40];
    const float* aow = (const float*)d_in[41]; const float* aob = (const float*)d_in[42];
    const float* s0w = (const float*)d_in[43]; const float* s0b = (const float*)d_in[44];
    const float* s1w = (const float*)d_in[45]; const float* s1b = (const float*)d_in[46];
    float* out = (float*)d_out;

    const int N = in_sizes[0] / 6;        // 50000
    const int E = in_sizes[1] / 2;        // 400000

    // ---- workspace layout (floats) ----
    float* ws = (float*)d_ws;
    const size_t NB = (size_t)N * 512;    // one big node buffer (== E*64)
    float* buf0 = ws;
    float* buf1 = ws + NB;
    float* buf2 = ws + 2 * NB;
    float* tl   = ws + 3 * NB;
    float* cnt  = tl;                 tl += N;
    float* gcnt = tl;                 tl += 256;
    float* gsum = tl;                 tl += 256 * 512;
    unsigned* gmaxb = (unsigned*)tl;  tl += 256 * 512;
    float* z  = tl;                   tl += 256 * 1024;
    float* t0 = tl;                   tl += 256 * 1024;
    float* t1 = tl;                   tl += 256 * 1024;
    float* ha0 = tl;                  tl += 256 * 64;
    float* ha1 = tl;                  tl += 256 * 64;
    float* hs0 = tl;                  tl += 256 * 64;
    float* hs1 = tl;                  tl += 256 * 64;
    float* w0p = tl;                  tl += 32 * 64;

    auto zero = [&](float* p, long long n) {
        zero_kernel<<<(unsigned)((n + 255) / 256), 256, 0, stream>>>(p, n);
    };
    auto gemm = [&](const float* A, int lda, const float* B, int ldb,
                    float* D, int ldd, int M, int Nn, int K,
                    const float* bias, int mode,
                    const float* g = nullptr, const float* bb = nullptr,
                    const float* mm = nullptr, const float* vv = nullptr) {
        dim3 grid((M / 16 + GW - 1) / GW, Nn / 32);
        wmma_gemm_kernel<<<grid, TB, 0, stream>>>(A, lda, B, ldb, D, ldd,
                                                  M, Nn, K, bias, mode,
                                                  g, bb, mm, vv);
    };

    // ---- ECC edge network ----
    zero(cnt, N);
    pad_w0_kernel<<<(32 * 64) / 256, 256, 0, stream>>>(w0, w0p);
    edge_feat_kernel<<<(E + 255) / 256, 256, 0, stream>>>(x, ei, buf0, cnt, E);
    gemm(buf0, 32, w0p, 64, buf1, 64, E, 64, 32, b0, 1);    // e -> k1 (relu)
    gemm(buf1, 64, w1, 64, buf2, 64, E, 64, 64, b1, 1);     // k1 -> k2 (relu)
    zero(buf0, (long long)N * 64);                          // h_sum
    {
        dim3 grid((E / 16 + GW - 1) / GW, 4);
        ecc_msg_kernel<<<grid, TB, 0, stream>>>(buf2, w2, b2, x, ei, buf0, E);
    }
    ecc_combine_kernel<<<(N * 64 + 255) / 256, 256, 0, stream>>>(
        buf0, cnt, x, root, rb, buf1, N);                   // h -> buf1 (N x 64)

    // ---- GCN layers: in -> hw -> agg (relu in place) ----
    auto gcn = [&](const float* hin, int cin, const float* W, const float* B,
                   float* hw, float* agg, int cout, int cshift) {
        gemm(hin, cin, W, cout, hw, cout, N, cout, cin, nullptr, 0);
        zero(agg, (long long)N << cshift);
        long long ne = (long long)E << cshift;
        gcn_scatter_kernel<<<(unsigned)((ne + 255) / 256), 256, 0, stream>>>(
            hw, ei, agg, E, cshift);
        long long nn = (long long)N << cshift;
        bias_relu_kernel<<<(unsigned)((nn + 255) / 256), 256, 0, stream>>>(
            agg, B, N, cshift);
    };
    gcn(buf1,  64, g1w, g1b, buf2, buf0,  64, 6);   // out buf0 (64)
    gcn(buf0,  64, g2w, g2b, buf1, buf2, 128, 7);   // out buf2 (128)
    gcn(buf2, 128, g3w, g3b, buf0, buf1, 256, 8);   // out buf1 (256)
    gcn(buf1, 256, g4w, g4b, buf2, buf0, 512, 9);   // out buf0 (512)

    // ---- pooling ----
    zero(gcnt, 256);
    zero(gsum, 256 * 512);
    zero((float*)gmaxb, 256 * 512);
    {
        long long nn = (long long)N << 9;
        pool_kernel<<<(unsigned)((nn + 255) / 256), 256, 0, stream>>>(
            buf0, gi, gsum, gmaxb, gcnt, N);
    }
    zbuild_kernel<<<(256 * 1024) / 256, 256, 0, stream>>>(gsum, gmaxb, gcnt, z);

    // ---- decoder (leaky-relu + BN epilogue) ----
    gemm(z,  1024, d0w, 1024, t0, 1024, 256, 1024, 1024, d0b, 2, bn0g, bn0b, bn0m, bn0v);
    gemm(t0, 1024, d1w, 1024, t1, 1024, 256, 1024, 1024, d1b, 2, bn1g, bn1b, bn1m, bn1v);
    gemm(t1, 1024, d2w,  512, t0,  512, 256,  512, 1024, d2b, 2, bn2g, bn2b, bn2m, bn2v);

    // ---- heads ----
    gemm(t0, 512, a0w, 64, ha0, 64, 256, 64, 512, a0b, 0);
    gemm(ha0, 64, a1w, 64, ha1, 64, 256, 64,  64, a1b, 0);
    gemm(t0, 512, s0w, 64, hs0, 64, 256, 64, 512, s0b, 0);
    gemm(hs0, 64, s1w, 64, hs1, 64, 256, 64,  64, s1b, 0);

    final_kernel<<<256, 64, 0, stream>>>(ha1, aow, aob, hs1, out);
}
